// InferModel_26886495273140
// MI455X (gfx1250) — matrix operand order repaired and checked by reading, not compiled, not run on hardware
//
#include <hip/hip_runtime.h>
#include <stdint.h>

// ---------------- problem constants (fixed by the reference) ----------------
#define S_NUM 4
#define E_NUM 8
#define D0 384
#define D1 160
#define D2 128
#define D3 96
#define KT1 12   // 384/32
#define NT1 10   // 160/16
#define KT2 5    // 160/32
#define NT2 8    // 128/16
#define KT3 4    // 128/32
#define NT3 6    // 96/16
#define FRAG_ELEMS 512           // one 32x16 bf16 B fragment (K x N)
#define FRAG_Q (FRAG_ELEMS/8)    // 64 uint4 per fragment
#define W1_FRAGS (KT1*NT1)       // 120
#define W2_FRAGS (KT2*NT2)       // 40
#define W3_FRAGS (KT3*NT3)       // 24
#define CELU_ALPHA 0.1f

// ---------------- WMMA types ----------------
typedef __attribute__((ext_vector_type(16))) __bf16 v16bf;
typedef __attribute__((ext_vector_type(8)))  float  v8f;

union Frag {
  v16bf v;
  uint4 q[2];
  unsigned short u[16];
};
static_assert(sizeof(Frag) == 32, "frag must be 8 VGPRs");

__device__ __forceinline__ unsigned short f32_bf16(float f) {
  union { __bf16 h; unsigned short u; } c;
  c.h = (__bf16)f;                 // native cvt; backend picks v_cvt path
  return c.u;
}
__device__ __forceinline__ float bf16_f32(unsigned short h) {
  union { unsigned int u; float f; } c; c.u = ((unsigned int)h) << 16;
  return c.f;
}
__device__ __forceinline__ float celu_f(float x) {
  return x > 0.f ? x : CELU_ALPHA * (__expf(x * (1.0f / CELU_ALPHA)) - 1.0f);
}

// ---------------- LDS layout (bytes) for ani_main ----------------
#define OFF_W1  0
#define OFF_W2  (OFF_W1 + W1_FRAGS*FRAG_ELEMS*2)   // 122880
#define OFF_W3  (OFF_W2 + W2_FRAGS*FRAG_ELEMS*2)   // 163840
#define OFF_B1  (OFF_W3 + W3_FRAGS*FRAG_ELEMS*2)   // 188416
#define OFF_B2  (OFF_B1 + D1*4)                    // 189056
#define OFF_B3  (OFF_B2 + D2*4)                    // 189568
#define OFF_W4  (OFF_B3 + D3*4)                    // 189952
#define OFF_B4  (OFF_W4 + D3*4)                    // 190336
#define OFF_ACC (OFF_B4 + 16)                      // 190352
#define OFF_BUF (OFF_ACC + 16)                     // 190368 (16B aligned)
#define SA1 168   // bufA row stride (holds up to 160 bf16) -> 336B rows
#define SB1 136   // bufB row stride (holds up to 128 bf16) -> 272B rows
#define BUF_PER_WAVE_E (16*(SA1+SB1))              // 4864 bf16 elems
#define LDS_TOTAL (OFF_BUF + 8*BUF_PER_WAVE_E*2)   // 268192 B  (< 320KB WGP LDS)

// ---------------- small prep kernels ----------------
__global__ void init_k(int* counts, float* out) {
  if (threadIdx.x < S_NUM) counts[threadIdx.x] = 0;
  if (threadIdx.x == 0) out[0] = 0.f;
}

__global__ void scatter_k(const int* __restrict__ species, int* __restrict__ counts,
                          int* __restrict__ idx, int N) {
  int i = blockIdx.x * blockDim.x + threadIdx.x;
  if (i >= N) return;
  int s = species[i];
  int pos = atomicAdd(&counts[s], 1);
  idx[(size_t)s * N + pos] = i;
}

// fp32 [net][K][Nn] -> bf16 swizzled [net][nt][kt][frag 512], frag lane-major:
// element (k,n): lane = (n&15) + 16*((k>>3)&1), slot = (k&7) + 8*((k&31)>=16)  (16 bf16/lane contiguous)
// [nt][kt] major order keeps the K-inner fragments 1KB apart -> DS offsets fit imm16.
__global__ void swizzle_w(const float* __restrict__ src, unsigned short* __restrict__ dst,
                          int K, int Nn, int NT, int total) {
  int i = blockIdx.x * blockDim.x + threadIdx.x;
  if (i >= total) return;
  int per = K * Nn;
  int net = i / per;
  int rem = i - net * per;
  int k = rem / Nn;
  int n = rem - k * Nn;
  int kt = k >> 5, kk = k & 31;
  int nt = n >> 4, nn = n & 15;
  int lane = nn + (((kk >> 3) & 1) << 4);
  int slot = (kk & 7) + ((kk >> 4) << 3);
  int KT = K >> 5;
  dst[((size_t)net * (KT * NT) + (size_t)(nt * KT + kt)) * FRAG_ELEMS + lane * 16 + slot] =
      f32_bf16(src[i]);
}

// ---------------- WMMA helpers ----------------
// A fragment per-lane layout (16x32 bf16, wave32):
//   lanes 0-15  : row M=lane,    elems 0..7 = K k0+0..7,  elems 8..15 = K k0+16..23
//   lanes 16-31 : row M=lane-16, elems 0..7 = K k0+8..15, elems 8..15 = K k0+24..31

// Dual-column sweep: two independent accumulator chains (nt, nt+1) for WMMA ILP,
// with software-pipelined (double-buffered) B fragments so ds_load latency hides
// under the matrix pipe instead of a full s_wait_dscnt 0 before every WMMA.
template <int KT, int NT>
__device__ __forceinline__ void wmma_sweep(const Frag* __restrict__ a,
                                           const unsigned short* __restrict__ Wlds,
                                           const float* __restrict__ biasLds,
                                           unsigned short* __restrict__ Dst, int strideD,
                                           int lane, bool act) {
  static_assert((NT & 1) == 0, "NT must be even");
  const int half = lane >> 4;
  const int l16 = lane & 15;
#pragma unroll 1
  for (int nt = 0; nt < NT; nt += 2) {
    const uint4* w0 = (const uint4*)(Wlds + (size_t)(nt)     * KT * FRAG_ELEMS) + lane * 2;
    const uint4* w1 = (const uint4*)(Wlds + (size_t)(nt + 1) * KT * FRAG_ELEMS) + lane * 2;
    v8f acc0 = {0.f, 0.f, 0.f, 0.f, 0.f, 0.f, 0.f, 0.f};
    v8f acc1 = {0.f, 0.f, 0.f, 0.f, 0.f, 0.f, 0.f, 0.f};
    Frag b0, b1, nb0, nb1;
    b0.q[0] = w0[0]; b0.q[1] = w0[1];
    b1.q[0] = w1[0]; b1.q[1] = w1[1];
#pragma unroll
    for (int kt = 0; kt < KT; ++kt) {
      if (kt + 1 < KT) {                       // prefetch next K-step's B frags
        const uint4* p0 = w0 + (kt + 1) * FRAG_Q;
        const uint4* p1 = w1 + (kt + 1) * FRAG_Q;
        nb0.q[0] = p0[0]; nb0.q[1] = p0[1];
        nb1.q[0] = p1[0]; nb1.q[1] = p1[1];
      }
      acc0 = __builtin_amdgcn_wmma_f32_16x16x32_bf16(false, a[kt].v, false, b0.v,
                                                     (short)0, acc0, false, false);
      acc1 = __builtin_amdgcn_wmma_f32_16x16x32_bf16(false, a[kt].v, false, b1.v,
                                                     (short)0, acc1, false, false);
      if (kt + 1 < KT) { b0 = nb0; b1 = nb1; }
    }
    const float bv0 = biasLds[nt * 16 + l16];
    const float bv1 = biasLds[(nt + 1) * 16 + l16];
#pragma unroll
    for (int r = 0; r < 8; ++r) {
      float x0 = acc0[r] + bv0;                // C layout: M = half*8 + r, N = l16
      float x1 = acc1[r] + bv1;
      if (act) { x0 = celu_f(x0); x1 = celu_f(x1); }
      Dst[(half * 8 + r) * strideD + nt * 16 + l16]       = f32_bf16(x0);
      Dst[(half * 8 + r) * strideD + (nt + 1) * 16 + l16] = f32_bf16(x1);
    }
  }
}

template <int KT, int NT>
__device__ __forceinline__ void layer_lds(const unsigned short* __restrict__ Asrc, int strideA,
                                          const unsigned short* __restrict__ Wlds,
                                          const float* __restrict__ biasLds,
                                          unsigned short* __restrict__ Dst, int strideD,
                                          int lane, bool act) {
  const int half = lane >> 4;
  const int l16 = lane & 15;
  Frag a[KT];
  const unsigned short* arow = Asrc + l16 * strideA + half * 8;
#pragma unroll
  for (int kt = 0; kt < KT; ++kt) {
    a[kt].q[0] = *(const uint4*)(arow + kt * 32);        // ds_load_b128
    a[kt].q[1] = *(const uint4*)(arow + kt * 32 + 16);
  }
  wmma_sweep<KT, NT>(a, Wlds, biasLds, Dst, strideD, lane, act);
}

// Layer 1: A gathered directly from global fp32 AEV rows, converted to bf16 in regs.
__device__ __forceinline__ void layer1_global(const float* __restrict__ aev,
                                              const int* __restrict__ rows, int nvalid,
                                              const unsigned short* __restrict__ Wlds,
                                              const float* __restrict__ biasLds,
                                              unsigned short* __restrict__ Dst, int strideD,
                                              int lane) {
  const int half = lane >> 4;
  const int l16 = lane & 15;
  const bool rv = (l16 < nvalid);
  Frag a[KT1];
  const float* src = aev;
  if (rv) src = aev + (long long)rows[l16] * D0 + half * 8;
#pragma unroll
  for (int kt = 0; kt < KT1; ++kt) {
    if (rv) {
      const float4* p0 = (const float4*)(src + kt * 32);       // K k0+half*8 .. +7
      const float4* p1 = (const float4*)(src + kt * 32 + 16);  // K +16 .. +23
      float4 c0 = p0[0], c1 = p0[1], c2 = p1[0], c3 = p1[1];
      a[kt].u[0] = f32_bf16(c0.x);  a[kt].u[1] = f32_bf16(c0.y);
      a[kt].u[2] = f32_bf16(c0.z);  a[kt].u[3] = f32_bf16(c0.w);
      a[kt].u[4] = f32_bf16(c1.x);  a[kt].u[5] = f32_bf16(c1.y);
      a[kt].u[6] = f32_bf16(c1.z);  a[kt].u[7] = f32_bf16(c1.w);
      a[kt].u[8]  = f32_bf16(c2.x); a[kt].u[9]  = f32_bf16(c2.y);
      a[kt].u[10] = f32_bf16(c2.z); a[kt].u[11] = f32_bf16(c2.w);
      a[kt].u[12] = f32_bf16(c3.x); a[kt].u[13] = f32_bf16(c3.y);
      a[kt].u[14] = f32_bf16(c3.z); a[kt].u[15] = f32_bf16(c3.w);
    } else {
      uint4 z = {0u, 0u, 0u, 0u};
      a[kt].q[0] = z;
      a[kt].q[1] = z;
    }
  }
  wmma_sweep<KT1, NT1>(a, Wlds, biasLds, Dst, strideD, lane, true);
}

// ---------------- main kernel: one block = one (species, ensemble) net ----------------
__global__ void __launch_bounds__(256, 1)
ani_main(const float* __restrict__ aev,
         const float* __restrict__ b1, const float* __restrict__ b2,
         const float* __restrict__ b3, const float* __restrict__ W4,
         const float* __restrict__ b4,
         const int* __restrict__ counts, const int* __restrict__ idx,
         const unsigned short* __restrict__ w1s, const unsigned short* __restrict__ w2s,
         const unsigned short* __restrict__ w3s,
         float* __restrict__ out, int N) {
  extern __shared__ char smem[];
  const int net  = blockIdx.y;          // 0..31 = species*8 + ensemble
  const int s    = net >> 3;
  const int tid  = threadIdx.x;
  const int wave = tid >> 5;
  const int lane = tid & 31;

  unsigned short* lw1 = (unsigned short*)(smem + OFF_W1);
  unsigned short* lw2 = (unsigned short*)(smem + OFF_W2);
  unsigned short* lw3 = (unsigned short*)(smem + OFF_W3);
  float* lb1  = (float*)(smem + OFF_B1);
  float* lb2  = (float*)(smem + OFF_B2);
  float* lb3  = (float*)(smem + OFF_B3);
  float* lw4  = (float*)(smem + OFF_W4);
  float* lb4  = (float*)(smem + OFF_B4);
  float* lacc = (float*)(smem + OFF_ACC);

  // ---- stage this net's weights into LDS once (≈188KB, reused for all atom tiles) ----
  {
    const uint4* g1 = (const uint4*)(w1s + (size_t)net * (W1_FRAGS * FRAG_ELEMS));
    uint4* l1 = (uint4*)lw1;
    for (int i = tid; i < W1_FRAGS * FRAG_ELEMS / 8; i += 256) l1[i] = g1[i];
    const uint4* g2 = (const uint4*)(w2s + (size_t)net * (W2_FRAGS * FRAG_ELEMS));
    uint4* l2 = (uint4*)lw2;
    for (int i = tid; i < W2_FRAGS * FRAG_ELEMS / 8; i += 256) l2[i] = g2[i];
    const uint4* g3 = (const uint4*)(w3s + (size_t)net * (W3_FRAGS * FRAG_ELEMS));
    uint4* l3 = (uint4*)lw3;
    for (int i = tid; i < W3_FRAGS * FRAG_ELEMS / 8; i += 256) l3[i] = g3[i];
    for (int i = tid; i < D1; i += 256) lb1[i] = b1[(size_t)net * D1 + i];
    for (int i = tid; i < D2; i += 256) lb2[i] = b2[(size_t)net * D2 + i];
    for (int i = tid; i < D3; i += 256) lb3[i] = b3[(size_t)net * D3 + i];
    for (int i = tid; i < D3; i += 256) lw4[i] = W4[(size_t)net * D3 + i];
    if (tid == 0) { lb4[0] = b4[net]; lacc[0] = 0.f; }
  }
  __syncthreads();

  const int cnt   = counts[s];
  const int tiles = (cnt + 15) >> 4;
  const int wid   = blockIdx.x * 8 + wave;   // 128 workers per net
  unsigned short* bufA = (unsigned short*)(smem + OFF_BUF) + (size_t)wave * BUF_PER_WAVE_E;
  unsigned short* bufB = bufA + 16 * SA1;

  float partial = 0.f;

  for (int t = wid; t < tiles; t += 128) {
    const int nvalid = min(16, cnt - t * 16);
    const int* rows = idx + (size_t)s * N + t * 16;

    // prefetch next tile's AEV rows (global_prefetch_b8)
    int tn = t + 128;
    if (tn < tiles) {
      const int* rn = idx + (size_t)s * N + tn * 16;
      int nv2 = min(16, cnt - tn * 16);
      if ((lane & 15) < nv2)
        __builtin_prefetch(aev + (long long)rn[lane & 15] * D0 + (lane >> 4) * (D0 / 2), 0, 1);
    }

    // L1: 16x384 @ 384x160 (+b1, CELU)  -> bufA
    layer1_global(aev, rows, nvalid, lw1, lb1, bufA, SA1, lane);
    // L2: 16x160 @ 160x128 (+b2, CELU)  -> bufB
    layer_lds<KT2, NT2>(bufA, SA1, lw2, lb2, bufB, SB1, lane, true);
    // L3: 16x128 @ 128x96  (+b3, CELU)  -> bufA
    layer_lds<KT3, NT3>(bufB, SB1, lw3, lb3, bufA, SA1, lane, true);

    // L4: 16x96 . 96 (+b4); lanes 0-15 each own one atom; ensemble mean = *0.125
    if (lane < nvalid) {
      float sum = lb4[0];
      const uint4* h4 = (const uint4*)(bufA + lane * SA1);   // 336B rows, 16B aligned
#pragma unroll
      for (int kq = 0; kq < D3 / 8; ++kq) {                  // 12 x ds_load_b128
        uint4 q = h4[kq];
        const float* w = lw4 + kq * 8;
        unsigned int uu[4] = {q.x, q.y, q.z, q.w};
#pragma unroll
        for (int j = 0; j < 4; ++j) {
          float flo = __uint_as_float(uu[j] << 16);
          float fhi = __uint_as_float(uu[j] & 0xffff0000u);
          sum += flo * w[2 * j] + fhi * w[2 * j + 1];
        }
      }
      partial += sum * 0.125f;
    }
  }

  // wave -> block -> global reduction
#pragma unroll
  for (int off = 16; off > 0; off >>= 1) partial += __shfl_xor(partial, off, 32);
  if (lane == 0) atomicAdd(lacc, partial);     // ds_add_f32
  __syncthreads();
  if (tid == 0) atomicAdd(out, lacc[0]);
}

// ---------------- host-side launch ----------------
extern "C" void kernel_launch(void* const* d_in, const int* in_sizes, int n_in,
                              void* d_out, int out_size, void* d_ws, size_t ws_size,
                              hipStream_t stream) {
  const float* aev     = (const float*)d_in[0];
  const int*   species = (const int*)d_in[1];
  const float* W1 = (const float*)d_in[2];
  const float* b1 = (const float*)d_in[3];
  const float* W2 = (const float*)d_in[4];
  const float* b2 = (const float*)d_in[5];
  const float* W3 = (const float*)d_in[6];
  const float* b3 = (const float*)d_in[7];
  const float* W4 = (const float*)d_in[8];
  const float* b4 = (const float*)d_in[9];
  float* out = (float*)d_out;
  const int N = in_sizes[1];

  // workspace layout: counters | per-species index buckets | swizzled bf16 W1/W2/W3
  char* ws = (char*)d_ws;
  int* counts = (int*)ws;
  int* idx    = (int*)(ws + 256);
  size_t off = 256 + (size_t)S_NUM * N * sizeof(int);
  off = (off + 255) & ~(size_t)255;
  unsigned short* w1s = (unsigned short*)(ws + off);
  off += (size_t)(S_NUM * E_NUM) * W1_FRAGS * FRAG_ELEMS * 2;
  unsigned short* w2s = (unsigned short*)(ws + off);
  off += (size_t)(S_NUM * E_NUM) * W2_FRAGS * FRAG_ELEMS * 2;
  unsigned short* w3s = (unsigned short*)(ws + off);

  init_k<<<1, 64, 0, stream>>>(counts, out);

  int t1 = S_NUM * E_NUM * D0 * D1;
  swizzle_w<<<(t1 + 255) / 256, 256, 0, stream>>>(W1, w1s, D0, D1, NT1, t1);
  int t2 = S_NUM * E_NUM * D1 * D2;
  swizzle_w<<<(t2 + 255) / 256, 256, 0, stream>>>(W2, w2s, D1, D2, NT2, t2);
  int t3 = S_NUM * E_NUM * D2 * D3;
  swizzle_w<<<(t3 + 255) / 256, 256, 0, stream>>>(W3, w3s, D2, D3, NT3, t3);

  scatter_k<<<(N + 255) / 256, 256, 0, stream>>>(species, counts, idx, N);

  dim3 grid(16, S_NUM * E_NUM);   // 16 chunks x 32 nets
  ani_main<<<grid, 256, LDS_TOTAL, stream>>>(aev, b1, b2, b3, W4, b4, counts, idx,
                                             w1s, w2s, w3s, out, N);
}
